// PAM_50689204027801
// MI455X (gfx1250) — compile-verified, matrix operand order checked by reading, not run on hardware
//
#include <hip/hip_runtime.h>

// Position Attention Module (DANet PAM), B=4, C=512, N=64*64=4096, CK=64.
// Flash-attention-style fused implementation for gfx1250 using bf16 WMMA
// (v_wmma_f32_16x16x32_bf16) with fp32 accumulation and fp32 softmax stats.

#define BATCH 4
#define CH    512
#define CKDIM 64
#define NPIX  4096

typedef __attribute__((ext_vector_type(16))) __bf16 v16bf;
typedef __attribute__((ext_vector_type(8)))  __bf16 v8bf;
typedef __attribute__((ext_vector_type(8)))  float  v8f;
typedef __attribute__((ext_vector_type(8)))  unsigned short u16x8;

static __device__ __forceinline__ unsigned short f2bf(float f) {
  unsigned u = __float_as_uint(f);
  u += 0x7fffu + ((u >> 16) & 1u);      // round-to-nearest-even
  return (unsigned short)(u >> 16);
}

// Build a 16x32 (or 32x16) bf16 WMMA fragment from two contiguous 16-byte
// chunks (K=0..7 / K=16..23 for lanes 0-15; K=8..15 / K=24..31 for lanes 16-31).
static __device__ __forceinline__ v16bf frag16(const unsigned short* p0,
                                               const unsigned short* p1) {
  v8bf lo = *reinterpret_cast<const v8bf*>(p0);
  v8bf hi = *reinterpret_cast<const v8bf*>(p1);
  return __builtin_shufflevector(lo, hi, 0,1,2,3,4,5,6,7,8,9,10,11,12,13,14,15);
}

static __device__ __forceinline__ v8f wmma_bf16(v16bf a, v16bf b, v8f c) {
  return __builtin_amdgcn_wmma_f32_16x16x32_bf16(false, a, false, b,
                                                 (short)0, c, false, false);
}

// ---------------------------------------------------------------- K0a: weights -> bf16
__global__ void convert_weights(const float* __restrict__ Wk,
                                const float* __restrict__ Wv,
                                unsigned short* __restrict__ Wkb,
                                unsigned short* __restrict__ Wvb) {
  int i = blockIdx.x * blockDim.x + threadIdx.x;
  if (i < CKDIM * CH) Wkb[i] = f2bf(Wk[i]);
  if (i < CH * CH)    Wvb[i] = f2bf(Wv[i]);
}

// ---------------------------------------------------------------- K0b: x[b][c][n] -> Xt[b][n][c] (bf16)
__global__ void transpose_x(const float* __restrict__ x,
                            unsigned short* __restrict__ Xt) {
  __shared__ float tile[32][33];
  int b  = blockIdx.z;
  int n0 = blockIdx.x * 32;
  int c0 = blockIdx.y * 32;
  const float* xb = x + (size_t)b * CH * NPIX;
  #pragma unroll
  for (int i = threadIdx.y; i < 32; i += 8)
    tile[i][threadIdx.x] = xb[(size_t)(c0 + i) * NPIX + n0 + threadIdx.x];
  __syncthreads();
  unsigned short* Xtb = Xt + (size_t)b * NPIX * CH;
  #pragma unroll
  for (int i = threadIdx.y; i < 32; i += 8)
    Xtb[(size_t)(n0 + i) * CH + c0 + threadIdx.x] = f2bf(tile[threadIdx.x][i]);
}

// ---------------------------------------------------------------- K1: projection GEMM
// Pt[n,o] = sum_c Xt[n,c] * W[o,c] + bias[o]
// mode 0 (Q): store Qt[b][n][o]   (o = ck, row-major, Odim=64)
// mode 1 (V): store Vb[b][o][n]   (m-contiguous, Odim=512) -- b128 store from D tile
__global__ void proj_gemm(const unsigned short* __restrict__ Xt,
                          const unsigned short* __restrict__ Wb,
                          const float* __restrict__ bias,
                          unsigned short* __restrict__ outp,
                          int mode) {
  int b    = blockIdx.z;
  int n0   = blockIdx.x * 16;
  int o0   = blockIdx.y * 64;
  int lane = threadIdx.x;
  int half = lane >> 4;
  int lr   = lane & 15;

  const unsigned short* XtRow = Xt + (size_t)b * NPIX * CH + (size_t)(n0 + lr) * CH;
  v8f acc[4] = {};
  for (int c0 = 0; c0 < CH; c0 += 32) {
    v16bf a = frag16(XtRow + c0 + half * 8, XtRow + c0 + 16 + half * 8);
    #pragma unroll
    for (int ot = 0; ot < 4; ot++) {
      const unsigned short* Wr = Wb + (size_t)(o0 + ot * 16 + lr) * CH;
      v16bf bm = frag16(Wr + c0 + half * 8, Wr + c0 + 16 + half * 8);
      acc[ot] = wmma_bf16(a, bm, acc[ot]);
    }
  }
  #pragma unroll
  for (int ot = 0; ot < 4; ot++) {
    int o = o0 + ot * 16 + lr;                 // N dim of D tile
    float bo = bias[o];
    if (mode == 0) {
      unsigned short* Qt = outp + (size_t)b * NPIX * CKDIM;
      #pragma unroll
      for (int v = 0; v < 8; v++) {
        int n = n0 + v + 8 * half;             // M dim of D tile
        Qt[(size_t)n * CKDIM + o] = f2bf(acc[ot][v] + bo);
      }
    } else {
      u16x8 pack;
      #pragma unroll
      for (int v = 0; v < 8; v++) pack[v] = f2bf(acc[ot][v] + bo);
      unsigned short* Vb = outp + (size_t)b * CH * NPIX;
      *reinterpret_cast<u16x8*>(Vb + (size_t)o * NPIX + n0 + 8 * half) = pack;
    }
  }
}

// ---------------------------------------------------------------- K2: flash pass 1 (row max / row sumexp)
// grid: (NPIX/64, BATCH), block 128 (4 waves, 16 query rows each)
__global__ void attn_stats(const unsigned short* __restrict__ Qt,
                           float* __restrict__ row_max,
                           float* __restrict__ row_sum) {
  int b    = blockIdx.y;
  int wave = threadIdx.x >> 5;
  int lane = threadIdx.x & 31;
  int half = lane >> 4;
  int lr   = lane & 15;
  int n0   = blockIdx.x * 64 + wave * 16;

  const unsigned short* QtB  = Qt + (size_t)b * NPIX * CKDIM;
  const unsigned short* qrow = QtB + (size_t)(n0 + lr) * CKDIM;
  v16bf a0 = frag16(qrow + half * 8,      qrow + 16 + half * 8);   // c 0..31
  v16bf a1 = frag16(qrow + 32 + half * 8, qrow + 48 + half * 8);   // c 32..63

  float m_run[8], l_run[8];
  #pragma unroll
  for (int v = 0; v < 8; v++) { m_run[v] = -1e30f; l_run[v] = 0.0f; }

  for (int m0 = 0; m0 < NPIX; m0 += 16) {
    const unsigned short* krow = QtB + (size_t)(m0 + lr) * CKDIM;
    v16bf b0 = frag16(krow + half * 8,      krow + 16 + half * 8);
    v16bf b1 = frag16(krow + 32 + half * 8, krow + 48 + half * 8);
    v8f s = {};
    s = wmma_bf16(a0, b0, s);
    s = wmma_bf16(a1, b1, s);
    #pragma unroll
    for (int v = 0; v < 8; v++) {
      float t = s[v];                          // tile row max over the 16 lanes of this half
      t = fmaxf(t, __shfl_xor(t, 1, 32));
      t = fmaxf(t, __shfl_xor(t, 2, 32));
      t = fmaxf(t, __shfl_xor(t, 4, 32));
      t = fmaxf(t, __shfl_xor(t, 8, 32));
      float mn = fmaxf(m_run[v], t);
      float e  = __expf(s[v] - mn);
      e += __shfl_xor(e, 1, 32);
      e += __shfl_xor(e, 2, 32);
      e += __shfl_xor(e, 4, 32);
      e += __shfl_xor(e, 8, 32);
      l_run[v] = l_run[v] * __expf(m_run[v] - mn) + e;
      m_run[v] = mn;
    }
  }
  if (lr == 0) {
    #pragma unroll
    for (int v = 0; v < 8; v++) {
      int n = n0 + v + 8 * half;
      row_max[(size_t)b * NPIX + n] = m_run[v];
      row_sum[(size_t)b * NPIX + n] = l_run[v];
    }
  }
}

// ---------------------------------------------------------------- K3: flash pass 2 (fused P*V^T + epilogue)
// grid: (NPIX/16, BATCH), block 128 (4 waves). WG owns 16 query rows x all 512 c.
// Per 128-wide m block: each wave builds a 16x32 P tile into LDS (D->A relayout),
// then consumes all 128 m against its private 128-wide c slice (8 c-tiles).
__global__ void attn_out(const unsigned short* __restrict__ Qt,
                         const unsigned short* __restrict__ Vb,
                         const float* __restrict__ row_max,
                         const float* __restrict__ row_sum,
                         const float* __restrict__ x,
                         const float* __restrict__ gamma,
                         float* __restrict__ out) {
  __shared__ unsigned short Plds[16][136];     // 128 m + 8 pad (bank skew)
  int b    = blockIdx.y;
  int n0   = blockIdx.x * 16;
  int wave = threadIdx.x >> 5;
  int lane = threadIdx.x & 31;
  int half = lane >> 4;
  int lr   = lane & 15;

  const unsigned short* QtB  = Qt + (size_t)b * NPIX * CKDIM;
  const unsigned short* VbB  = Vb + (size_t)b * CH * NPIX;
  const unsigned short* qrow = QtB + (size_t)(n0 + lr) * CKDIM;
  v16bf a0 = frag16(qrow + half * 8,      qrow + 16 + half * 8);
  v16bf a1 = frag16(qrow + 32 + half * 8, qrow + 48 + half * 8);

  float mrow[8], isrow[8];
  #pragma unroll
  for (int v = 0; v < 8; v++) {
    int n = n0 + v + 8 * half;
    mrow[v]  = row_max[(size_t)b * NPIX + n];
    isrow[v] = 1.0f / row_sum[(size_t)b * NPIX + n];
  }

  v8f acc[8] = {};
  for (int mblk = 0; mblk < NPIX; mblk += 128) {
    // phase 1: this wave computes P for m in [mblk + 32*wave, +32)
    #pragma unroll
    for (int j = 0; j < 2; j++) {
      int m0 = mblk + wave * 32 + j * 16;
      const unsigned short* krow = QtB + (size_t)(m0 + lr) * CKDIM;
      v16bf b0 = frag16(krow + half * 8,      krow + 16 + half * 8);
      v16bf b1 = frag16(krow + 32 + half * 8, krow + 48 + half * 8);
      v8f s = {};
      s = wmma_bf16(a0, b0, s);
      s = wmma_bf16(a1, b1, s);
      #pragma unroll
      for (int v = 0; v < 8; v++) {
        float p = __expf(s[v] - mrow[v]) * isrow[v];
        Plds[v + 8 * half][wave * 32 + j * 16 + lr] = f2bf(p);
      }
    }
    __syncthreads();
    // phase 2: consume 128 m against c slice [128*wave, 128*wave+128)
    #pragma unroll
    for (int kk = 0; kk < 4; kk++) {
      const unsigned short* prow = &Plds[lr][kk * 32];
      v16bf ap = frag16(prow + half * 8, prow + 16 + half * 8);
      #pragma unroll
      for (int ct = 0; ct < 8; ct++) {
        int c = wave * 128 + ct * 16 + lr;
        const unsigned short* vrow = VbB + (size_t)c * NPIX + mblk + kk * 32;
        v16bf bv = frag16(vrow + half * 8, vrow + 16 + half * 8);
        acc[ct] = wmma_bf16(ap, bv, acc[ct]);
      }
    }
    __syncthreads();
  }

  float g = gamma[0];
  const float* xB  = x  + (size_t)b * CH * NPIX;
  float*       oB  = out + (size_t)b * CH * NPIX;
  #pragma unroll
  for (int ct = 0; ct < 8; ct++) {
    int c = wave * 128 + ct * 16 + lr;
    #pragma unroll
    for (int v = 0; v < 8; v++) {
      int n = n0 + v + 8 * half;
      size_t idx = (size_t)c * NPIX + n;
      oB[idx] = g * acc[ct][v] + xB[idx];
    }
  }
}

// ---------------------------------------------------------------- host
extern "C" void kernel_launch(void* const* d_in, const int* in_sizes, int n_in,
                              void* d_out, int out_size, void* d_ws, size_t ws_size,
                              hipStream_t stream) {
  (void)in_sizes; (void)n_in; (void)out_size; (void)ws_size;
  const float* x     = (const float*)d_in[0];
  const float* Wk    = (const float*)d_in[1];
  const float* bk    = (const float*)d_in[2];
  const float* Wv    = (const float*)d_in[3];
  const float* bv    = (const float*)d_in[4];
  const float* gamma = (const float*)d_in[5];
  float* out = (float*)d_out;

  // Workspace carve (~36.5 MB total, bf16 intermediates + fp32 stats)
  unsigned short* Xt  = (unsigned short*)d_ws;                    // [B][N][C]
  unsigned short* Vb  = Xt  + (size_t)BATCH * NPIX * CH;          // [B][C][N]
  unsigned short* Qt  = Vb  + (size_t)BATCH * CH * NPIX;          // [B][N][CK]
  unsigned short* Wkb = Qt  + (size_t)BATCH * NPIX * CKDIM;       // [CK][C]
  unsigned short* Wvb = Wkb + (size_t)CKDIM * CH;                 // [C][C]
  float* row_max = (float*)(Wvb + (size_t)CH * CH);               // [B][N]
  float* row_sum = row_max + (size_t)BATCH * NPIX;                // [B][N]

  convert_weights<<<dim3((CH * CH + 255) / 256), 256, 0, stream>>>(Wk, Wv, Wkb, Wvb);
  transpose_x<<<dim3(NPIX / 32, CH / 32, BATCH), dim3(32, 8), 0, stream>>>(x, Xt);
  proj_gemm<<<dim3(NPIX / 16, CKDIM / 64, BATCH), 32, 0, stream>>>(Xt, Wkb, bk, Qt, 0);
  proj_gemm<<<dim3(NPIX / 16, CH / 64,    BATCH), 32, 0, stream>>>(Xt, Wvb, bv, Vb, 1);
  attn_stats<<<dim3(NPIX / 64, BATCH), 128, 0, stream>>>(Qt, row_max, row_sum);
  attn_out<<<dim3(NPIX / 16, BATCH), 128, 0, stream>>>(Qt, Vb, row_max, row_sum, x, gamma, out);
}
